// Hyper_GRU_22368189678088
// MI455X (gfx1250) — matrix operand order repaired and checked by reading, not compiled, not run on hardware
//
#include <hip/hip_runtime.h>
#include <cmath>

// ---------------------------------------------------------------------------
// Hyper_GRU for MI455X (gfx1250, wave32, WMMA 16x16x32 f16 -> f32)
//   pad/gather -> p2h MHA -> a2p MHA -> maxpool h0 -> bidirectional GRU ->
//   masked mean pool -> [512, 512] f32
// GEMM: two 16-row A strips resident per wave (128 VGPRs) so every B-tile
// load feeds two WMMAs; residual fused via template (no per-element branch).
// ---------------------------------------------------------------------------

typedef _Float16 half_t;
typedef __attribute__((ext_vector_type(16))) _Float16 v16h;
typedef __attribute__((ext_vector_type(8)))  _Float16 v8h;
typedef __attribute__((ext_vector_type(8)))  float    v8f;

#define WMMA_F16(a, b, c) \
  __builtin_amdgcn_wmma_f32_16x16x32_f16(false, (a), false, (b), (short)0, (c), false, false)

// ---- geometry -------------------------------------------------------------
constexpr int B_   = 512;
constexpr int H_   = 256;
constexpr int TA_  = 200;   // MAX_ATOM
constexpr int TP_  = 40;    // MAX_PHARM
constexpr int TH2_ = 80;    // MAX_HYPER

// ---- workspace layout (bytes) --------------------------------------------
constexpr size_t OFF_STARTS = 0;                                   // 3*512 int
constexpr size_t OFF_W16    = 8192;                                // attn weights f16
constexpr size_t W16A_B     = 8ull * 65536 * 2;                    // 8 x 256x256
constexpr size_t OFF_GRUW   = OFF_W16 + W16A_B;                    // 4 x 768x256 f16
constexpr size_t GRUW_B     = 4ull * 196608 * 2;
constexpr size_t OFF_ATOMF  = OFF_GRUW + GRUW_B;                   // [512,200,256] f16
constexpr size_t ATOMF_B    = (size_t)B_ * TA_ * H_ * 2;
constexpr size_t OFF_PHARMF = OFF_ATOMF + ATOMF_B;                 // [512,40,256] f16
constexpr size_t PHARMF_B   = (size_t)B_ * TP_ * H_ * 2;
constexpr size_t OFF_HYPERF = OFF_PHARMF + PHARMF_B;               // [512,80,256] f16
constexpr size_t HYPERF_B   = (size_t)B_ * TH2_ * H_ * 2;
constexpr size_t OFF_Q      = OFF_HYPERF + HYPERF_B;               // Q buf / GRU-x alias
constexpr size_t OFF_K      = OFF_Q + ATOMF_B;                     // K buf
constexpr size_t OFF_V      = OFF_K + HYPERF_B;                    // V buf
constexpr size_t OFF_X      = OFF_V + HYPERF_B;                    // attn out pre-Wo
constexpr size_t OFF_H0     = OFF_X + ATOMF_B;                     // [512,256] f16
// total ~234 MB

// ---- fragment loaders (ISA 7.12.2 layouts) --------------------------------
__device__ __forceinline__ v16h load_a_frag(const half_t* row0, int ld, int lo, int hi) {
  // A 16x32 f16: lane row M = lo; elements i -> K = i + (i&8) + 8*hi
  const half_t* p = row0 + (size_t)lo * ld + 8 * hi;
  v8h c0 = *(const v8h*)(p);
  v8h c1 = *(const v8h*)(p + 16);
  v16h a;
  for (int i = 0; i < 8; i++) { a[i] = c0[i]; a[i + 8] = c1[i]; }
  return a;
}
__device__ __forceinline__ v16h load_b_frag(const half_t* col0, int ld, int lo, int hi) {
  // B 32x16 f16: lane column N = lo; elements i -> K = i + (i&8) + 8*hi; B[k][n] = W[n][k]
  const half_t* p = col0 + (size_t)lo * ld + 8 * hi;
  v8h c0 = *(const v8h*)(p);
  v8h c1 = *(const v8h*)(p + 16);
  v16h b;
  for (int i = 0; i < 8; i++) { b[i] = c0[i]; b[i + 8] = c1[i]; }
  return b;
}

// ---- prefix-sum of counts -------------------------------------------------
__global__ void scan_kernel(const int* ac, const int* pc, const int* hc,
                            int* sa, int* sp, int* sh) {
  int t = threadIdx.x;
  if (t == 0) { int s = 0; for (int i = 0; i < B_; i++) { sa[i] = s; s += ac[i]; } }
  if (t == 1) { int s = 0; for (int i = 0; i < B_; i++) { sp[i] = s; s += pc[i]; } }
  if (t == 2) { int s = 0; for (int i = 0; i < B_; i++) { sh[i] = s; s += hc[i]; } }
}

// ---- gather ragged messages into zero-padded f16 [B, maxN, 256] -----------
__global__ void pad_kernel(const float* __restrict__ msg, const int* __restrict__ counts,
                           const int* __restrict__ starts, half_t* __restrict__ dst,
                           int maxN) {
  int row = blockIdx.x;              // b*maxN + n
  int b = row / maxN, n = row - b * maxN;
  int j = threadIdx.x;               // 256 threads = H
  float v = 0.f;
  if (n < counts[b]) v = msg[(size_t)(starts[b] + n) * H_ + j];
  dst[(size_t)row * H_ + j] = (half_t)v;
}

// ---- f32 -> f16 weight conversion -----------------------------------------
__global__ void cvt_kernel(const float* __restrict__ src, half_t* __restrict__ dst, int n) {
  int i = blockIdx.x * 256 + threadIdx.x;
  if (i < n) dst[i] = (half_t)src[i];
}

// ---- Y[rows,256] = X[rows,256](f16) @ W[256,256]^T(f16) + bias (+res) -----
// Two 16-row strips per wave: A fragments resident, each B tile -> 2 WMMAs.
template <bool HAS_RES>
__launch_bounds__(256)
__global__ void gemm_kernel(const half_t* __restrict__ X, const half_t* __restrict__ W,
                            const float* __restrict__ bias, const half_t* __restrict__ res,
                            half_t* __restrict__ Y, int rows) {
  int wave = threadIdx.x >> 5, lane = threadIdx.x & 31;
  int lo = lane & 15, hi = (lane >> 4) & 1;
  int rt0 = (blockIdx.x * 8 + wave) * 2;        // first of two row tiles
  if (rt0 >= (rows >> 4)) return;
  const half_t* X0 = X + (size_t)rt0 * 16 * H_;
  const half_t* X1 = X0 + 16 * H_;
  v16h a0[8], a1[8];
  for (int ks = 0; ks < 8; ks++) {
    a0[ks] = load_a_frag(X0 + ks * 32, H_, lo, hi);
    a1[ks] = load_a_frag(X1 + ks * 32, H_, lo, hi);
  }
  for (int ct = 0; ct < 16; ct++) {
    const half_t* Wb = W + (size_t)ct * 16 * H_;
    v8f c0 = {}, c1 = {};
    for (int ks = 0; ks < 8; ks++) {
      v16h bf = load_b_frag(Wb + ks * 32, H_, lo, hi);
      c0 = WMMA_F16(a0[ks], bf, c0);
      c1 = WMMA_F16(a1[ks], bf, c1);
    }
    float bc = bias[ct * 16 + lo];
    for (int r = 0; r < 8; r++) {
      size_t row0 = (size_t)rt0 * 16 + r + 8 * hi;
      size_t off0 = row0 * H_ + ct * 16 + lo;
      size_t off1 = off0 + 16 * H_;
      float v0 = c0[r] + bc, v1 = c1[r] + bc;
      if (HAS_RES) { v0 += (float)res[off0]; v1 += (float)res[off1]; }
      Y[off0] = (half_t)v0;
      Y[off1] = (half_t)v1;
    }
  }
}

// ---- masked MHA core: one block per (graph, head) -------------------------
__launch_bounds__(256)
__global__ void attn_kernel(const half_t* __restrict__ Q, const half_t* __restrict__ Kg,
                            const half_t* __restrict__ Vg, half_t* __restrict__ Xo,
                            const int* __restrict__ qcnt, const int* __restrict__ kcnt,
                            int Lq, int Lk, int LkP, int LqT) {
  __shared__ half_t Kl[96 * 64];       // [keyRow][d]
  __shared__ half_t Vt[64 * 96];       // [d][keyRow]  (transposed V)
  __shared__ float  Ss[8][16 * 96];    // per-wave scores
  __shared__ half_t Ps[8][16 * 96];    // per-wave probabilities
  int b = blockIdx.x >> 2, head = blockIdx.x & 3, hb = head * 64;
  int tid = threadIdx.x, wave = tid >> 5, lane = tid & 31;
  int lo = lane & 15, hi = (lane >> 4) & 1;

  // stage K head-slice and transposed V head-slice (zero-pad rows >= Lk)
  int tot = LkP * 64;
  for (int idx = tid; idx < tot; idx += 256) {
    int kk = idx >> 6, d = idx & 63;
    half_t kv = (half_t)0.f, vv = (half_t)0.f;
    if (kk < Lk) {
      size_t roff = (size_t)(b * Lk + kk) * H_ + hb + d;
      kv = Kg[roff]; vv = Vg[roff];
    }
    Kl[kk * 64 + d] = kv;
    Vt[d * 96 + kk] = vv;
  }
  __syncthreads();

  int qc = qcnt[b], kc = kcnt[b];
  float*  S = Ss[wave];
  half_t* P = Ps[wave];

  for (int qt = wave; qt < LqT; qt += 8) {
    int q0 = qt * 16;
    // ---- scores S = Q K^T : Q fragments loaded once ----
    int qr = q0 + lo; if (qr > Lq - 1) qr = Lq - 1;   // clamp (stores guarded)
    const half_t* arow = Q + (size_t)(b * Lq + qr) * H_ + hb;
    v16h qa[2];
    for (int ks = 0; ks < 2; ks++) qa[ks] = load_a_frag(arow + ks * 32, H_, lo, hi);
    int kTiles = LkP >> 4;
    for (int kt = 0; kt < kTiles; kt++) {
      v8f c = {};
      for (int ks = 0; ks < 2; ks++)
        c = WMMA_F16(qa[ks],
                     load_b_frag(Kl + (size_t)(kt * 16) * 64 + ks * 32, 64, lo, hi), c);
      for (int r = 0; r < 8; r++) S[(r + 8 * hi) * 96 + kt * 16 + lo] = c[r];
    }
    // ---- masked softmax, one lane per query row ----
    if (lane < 16) {
      int row = lane, qg = q0 + row;
      float m = -3.4e38f;
      for (int k = 0; k < LkP; k++) {
        float s = S[row * 96 + k];
        if (k >= Lk)                  s = -__builtin_inff();  // pad column
        else if (k >= kc || qg >= qc) s = -1e9f;              // reference mask fill
        else                          s *= 0.125f;            // 1/sqrt(64)
        S[row * 96 + k] = s;
        if (s > m) m = s;
      }
      float sum = 0.f;
      for (int k = 0; k < LkP; k++) {
        float e = expf(S[row * 96 + k] - m);
        S[row * 96 + k] = e; sum += e;
      }
      float inv = 1.0f / sum;
      for (int k = 0; k < LkP; k++) P[row * 96 + k] = (half_t)(S[row * 96 + k] * inv);
    }
    // ---- out = P V : P fragments loaded once ----
    int pks = LkP >> 5;
    v16h pa[3];
    for (int ks = 0; ks < pks; ks++) pa[ks] = load_a_frag(P + ks * 32, 96, lo, hi);
    for (int ct = 0; ct < 4; ct++) {
      v8f c = {};
      for (int ks = 0; ks < pks; ks++)
        c = WMMA_F16(pa[ks],
                     load_b_frag(Vt + (size_t)(ct * 16) * 96 + ks * 32, 96, lo, hi), c);
      for (int r = 0; r < 8; r++) {
        int q = q0 + r + 8 * hi;
        if (q < Lq)
          Xo[(size_t)(b * Lq + q) * H_ + hb + ct * 16 + lo] = (half_t)c[r];
      }
    }
  }
}

// ---- h0 = max over T ------------------------------------------------------
__global__ void maxpool_kernel(const half_t* __restrict__ hbuf, half_t* __restrict__ h0) {
  int b = blockIdx.x, j = threadIdx.x;
  float m = -3.4e38f;
  for (int t = 0; t < TA_; t++)
    m = fmaxf(m, (float)hbuf[((size_t)b * TA_ + t) * H_ + j]);
  h0[(size_t)b * H_ + j] = (half_t)m;
}

// ---- one GRU direction: 16 graphs per block, 200 steps on-chip ------------
// Waves 0,2,4,6 -> gi (x @ Wih^T); waves 1,3,5,7 -> gh (h @ Whh^T).
// Each wave loads its 8 A fragments from LDS once per step, sweeps 12 tiles.
__launch_bounds__(256)
__global__ void gru_kernel(const half_t* __restrict__ x, const half_t* __restrict__ h0,
                           const half_t* __restrict__ Wih, const half_t* __restrict__ Whh,
                           const float* __restrict__ bih, const float* __restrict__ bhh,
                           const int* __restrict__ counts, float* __restrict__ out,
                           int colOff, int rev) {
  __shared__ half_t hS[16 * 256];      // hidden state (f16, A-operand)
  __shared__ half_t xS[16 * 256];      // current x_t
  __shared__ float  giS[16 * 768];     // input-gate projections
  __shared__ float  ghS[16 * 768];     // hidden-gate projections
  __shared__ float  acc[16 * 256];     // masked sum for mean pool
  int g0 = blockIdx.x * 16;
  int tid = threadIdx.x, wave = tid >> 5, lane = tid & 31;
  int lo = lane & 15, hi = (lane >> 4) & 1;

  int mat = wave & 1;                  // 0: gi from xS, 1: gh from hS
  int cbase = wave >> 1;               // 0..3
  const half_t* Wm   = mat ? Whh : Wih;
  const half_t* src  = mat ? hS  : xS;
  float*       dst   = mat ? ghS : giS;
  const float* bias  = mat ? bhh : bih;

  for (int idx = tid; idx < 4096; idx += 256) {
    int m = idx >> 8;
    hS[idx] = h0[(size_t)(g0 + m) * H_ + (idx & 255)];
    acc[idx] = 0.f;
  }
  for (int t = 0; t < TA_; t++) {
    int tt = rev ? (TA_ - 1 - t) : t;
    for (int idx = tid; idx < 4096; idx += 256) {
      int m = idx >> 8;
      xS[idx] = x[((size_t)(g0 + m) * TA_ + tt) * H_ + (idx & 255)];
    }
    __syncthreads();
    // A fragments once per step (from LDS), then 12 column tiles of 768
    {
      const half_t* srow = src + (size_t)lo * H_;
      v16h a[8];
      for (int ks = 0; ks < 8; ks++) {
        const half_t* ap = srow + ks * 32 + 8 * hi;
        v8h a0 = *(const v8h*)ap;
        v8h a1 = *(const v8h*)(ap + 16);
        v16h av; for (int i = 0; i < 8; i++) { av[i] = a0[i]; av[i + 8] = a1[i]; }
        a[ks] = av;
      }
      for (int j = 0; j < 12; j++) {
        int ct = cbase + 4 * j;                      // 0..47, per-mat coverage
        const half_t* wrow = Wm + ((size_t)(ct * 16) + lo) * H_;
        v8f c = {};
        for (int ks = 0; ks < 8; ks++)
          c = WMMA_F16(a[ks], load_b_frag(wrow + ks * 32, H_, 0, hi), c);
        float bc = bias[ct * 16 + lo];
        for (int r = 0; r < 8; r++)
          dst[(r + 8 * hi) * 768 + ct * 16 + lo] = c[r] + bc;
      }
    }
    __syncthreads();
    // gates + state update + masked accumulation
    for (int idx = tid; idx < 4096; idx += 256) {
      int m = idx >> 8, j = idx & 255;
      float ir = giS[m * 768 + j],        hr = ghS[m * 768 + j];
      float iz = giS[m * 768 + 256 + j],  hz = ghS[m * 768 + 256 + j];
      float in_ = giS[m * 768 + 512 + j], hn = ghS[m * 768 + 512 + j];
      float r = 1.f / (1.f + expf(-(ir + hr)));
      float z = 1.f / (1.f + expf(-(iz + hz)));
      float n = tanhf(in_ + r * hn);
      float hp = (float)hS[idx];
      float h2 = (1.f - z) * n + z * hp;
      hS[idx] = (half_t)h2;
      if (tt < counts[g0 + m]) acc[idx] += h2;
    }
    __syncthreads();
  }
  for (int idx = tid; idx < 4096; idx += 256) {
    int m = idx >> 8, j = idx & 255;
    out[(size_t)(g0 + m) * 512 + colOff + j] = acc[idx] / (float)counts[g0 + m];
  }
}

// ---------------------------------------------------------------------------
extern "C" void kernel_launch(void* const* d_in, const int* in_sizes, int n_in,
                              void* d_out, int out_size, void* d_ws, size_t ws_size,
                              hipStream_t stream) {
  (void)in_sizes; (void)n_in; (void)out_size; (void)ws_size;
  const float* atom_msg  = (const float*)d_in[0];
  const float* pharm_msg = (const float*)d_in[1];
  const float* hyper_msg = (const float*)d_in[2];
  const int* acnt = (const int*)d_in[3];
  const int* pcnt = (const int*)d_in[4];
  const int* hcnt = (const int*)d_in[5];

  char* ws = (char*)d_ws;
  int* sa = (int*)(ws + OFF_STARTS);
  int* sp = sa + 512;
  int* sh = sa + 1024;
  half_t* w16    = (half_t*)(ws + OFF_W16);
  half_t* gw     = (half_t*)(ws + OFF_GRUW);
  half_t* atomF  = (half_t*)(ws + OFF_ATOMF);
  half_t* pharmF = (half_t*)(ws + OFF_PHARMF);
  half_t* hyperF = (half_t*)(ws + OFF_HYPERF);
  half_t* Qb     = (half_t*)(ws + OFF_Q);
  half_t* Kb     = (half_t*)(ws + OFF_K);
  half_t* Vb     = (half_t*)(ws + OFF_V);
  half_t* Xb     = (half_t*)(ws + OFF_X);
  half_t* h0b    = (half_t*)(ws + OFF_H0);
  half_t* hB     = Qb;                 // GRU input aliases Q buffer
  float* out = (float*)d_out;

  // 1) ragged -> padded f16
  scan_kernel<<<1, 32, 0, stream>>>(acnt, pcnt, hcnt, sa, sp, sh);
  pad_kernel<<<B_ * TA_,  256, 0, stream>>>(atom_msg,  acnt, sa, atomF,  TA_);
  pad_kernel<<<B_ * TP_,  256, 0, stream>>>(pharm_msg, pcnt, sp, pharmF, TP_);
  pad_kernel<<<B_ * TH2_, 256, 0, stream>>>(hyper_msg, hcnt, sh, hyperF, TH2_);

  // 2) weight conversion to f16
  static const int widx[8] = {6, 8, 10, 12, 14, 16, 18, 20};   // p2h Wq,Wk,Wv,Wo, a2p ...
  for (int i = 0; i < 8; i++)
    cvt_kernel<<<65536 / 256, 256, 0, stream>>>((const float*)d_in[widx[i]],
                                                w16 + (size_t)i * 65536, 65536);
  static const int gidx[4] = {22, 23, 26, 27};                 // Wih_f, Whh_f, Wih_b, Whh_b
  for (int i = 0; i < 4; i++)
    cvt_kernel<<<196608 / 256, 256, 0, stream>>>((const float*)d_in[gidx[i]],
                                                 gw + (size_t)i * 196608, 196608);

  auto gemm = [&](const half_t* X, const half_t* W, const float* bias,
                  half_t* Y, int rows) {
    gemm_kernel<false><<<rows / 256, 256, 0, stream>>>(X, W, bias, nullptr, Y, rows);
  };
  auto gemm_res = [&](const half_t* X, const half_t* W, const float* bias,
                      const half_t* res, half_t* Y, int rows) {
    gemm_kernel<true><<<rows / 256, 256, 0, stream>>>(X, W, bias, res, Y, rows);
  };

  // 3) p2h MHA: pharm queries attend hyper keys/values (+ residual into pharmF)
  gemm(pharmF, w16 + 0 * 65536, (const float*)d_in[7],  Qb, B_ * TP_);
  gemm(hyperF, w16 + 1 * 65536, (const float*)d_in[9],  Kb, B_ * TH2_);
  gemm(hyperF, w16 + 2 * 65536, (const float*)d_in[11], Vb, B_ * TH2_);
  attn_kernel<<<B_ * 4, 256, 0, stream>>>(Qb, Kb, Vb, Xb, pcnt, hcnt,
                                          TP_, TH2_, 96, 3);
  gemm_res(Xb, w16 + 3 * 65536, (const float*)d_in[13], pharmF, pharmF, B_ * TP_);

  // 4) a2p MHA: atom queries attend pharm keys/values (+ residual atomF -> hB)
  gemm(atomF,  w16 + 4 * 65536, (const float*)d_in[15], Qb, B_ * TA_);
  gemm(pharmF, w16 + 5 * 65536, (const float*)d_in[17], Kb, B_ * TP_);
  gemm(pharmF, w16 + 6 * 65536, (const float*)d_in[19], Vb, B_ * TP_);
  attn_kernel<<<B_ * 4, 256, 0, stream>>>(Qb, Kb, Vb, Xb, acnt, pcnt,
                                          TA_, TP_, 64, 13);
  gemm_res(Xb, w16 + 7 * 65536, (const float*)d_in[21], atomF, hB, B_ * TA_);

  // 5) h0 max-pool
  maxpool_kernel<<<B_, 256, 0, stream>>>(hB, h0b);

  // 6) bidirectional GRU + masked mean pool directly into d_out
  gru_kernel<<<B_ / 16, 256, 0, stream>>>(hB, h0b, gw, gw + 196608,
                                          (const float*)d_in[24], (const float*)d_in[25],
                                          acnt, out, 0,   0);
  gru_kernel<<<B_ / 16, 256, 0, stream>>>(hB, h0b, gw + 2 * 196608, gw + 3 * 196608,
                                          (const float*)d_in[28], (const float*)d_in[29],
                                          acnt, out, 256, 1);
}
